// SequenceContextualizer_22325240005463
// MI455X (gfx1250) — matrix operand order repaired and checked by reading, not compiled, not run on hardware
//
#include <hip/hip_runtime.h>
#include <hip/hip_bf16.h>

// ---------------------------------------------------------------------------
// Types for CDNA5 WMMA (gfx1250, wave32)
// ---------------------------------------------------------------------------
typedef __bf16 bf16;
typedef __attribute__((ext_vector_type(16))) __bf16 v16bf;
typedef __attribute__((ext_vector_type(8)))  float  v8f;

#define D_MODEL 128
#define N_HEADS 4
#define HEAD_D  32
#define SEG     256
#define FF_DIM  512
#define N_LAYERS 3

__device__ __forceinline__ bf16 f2bf(float f) {
  unsigned u = __builtin_bit_cast(unsigned, f);
  u += 0x7FFFu + ((u >> 16) & 1u);          // round-to-nearest-even
  unsigned short s = (unsigned short)(u >> 16);
  return __builtin_bit_cast(bf16, s);
}

// ---------------------------------------------------------------------------
// Embedding: x[n,:] = tok_emb[toks[n]] + pos_emb[n % 256] + mask_emb[tok==mask]
// one thread per 4 columns
// ---------------------------------------------------------------------------
__global__ __launch_bounds__(256) void embed_kernel(
    const int* __restrict__ toks, const int* __restrict__ mask_id_p,
    const float* __restrict__ tok_emb, const float* __restrict__ pos_emb,
    const float* __restrict__ mask_emb, float* __restrict__ x, int N)
{
  long idx = (long)blockIdx.x * blockDim.x + threadIdx.x;
  long n = idx >> 5;
  int  c = (int)(idx & 31) * 4;
  if (n >= N) return;
  int t = toks[n];
  int p = (int)(n & (SEG - 1));
  int mi = (t == *mask_id_p) ? 1 : 0;
  const float4 a = *(const float4*)(tok_emb + (long)t * D_MODEL + c);
  const float4 b = *(const float4*)(pos_emb + (long)p * D_MODEL + c);
  const float4 m = *(const float4*)(mask_emb + (long)mi * D_MODEL + c);
  float4 r; r.x = a.x + b.x + m.x; r.y = a.y + b.y + m.y;
  r.z = a.z + b.z + m.z; r.w = a.w + b.w + m.w;
  *(float4*)(x + n * D_MODEL + c) = r;
}

// ---------------------------------------------------------------------------
// fp32 -> bf16 conversion (for weights)
// ---------------------------------------------------------------------------
__global__ __launch_bounds__(256) void cvt_bf16_kernel(
    const float* __restrict__ in, bf16* __restrict__ out, int n)
{
  int i = blockIdx.x * blockDim.x + threadIdx.x;
  if (i < n) out[i] = f2bf(in[i]);
}

// ---------------------------------------------------------------------------
// LayerNorm over 128 cols: one wave32 per row, output bf16
// ---------------------------------------------------------------------------
__global__ __launch_bounds__(256) void ln_kernel(
    const float* __restrict__ x, const float* __restrict__ sc,
    const float* __restrict__ bi, bf16* __restrict__ y, int N)
{
  int wave = threadIdx.x >> 5, lane = threadIdx.x & 31;
  long row = (long)blockIdx.x * 8 + wave;
  if (row >= N) return;
  const float4 xv = *(const float4*)(x + row * D_MODEL + lane * 4);
  float sum = xv.x + xv.y + xv.z + xv.w;
  float sq  = xv.x * xv.x + xv.y * xv.y + xv.z * xv.z + xv.w * xv.w;
  #pragma unroll
  for (int off = 16; off > 0; off >>= 1) {
    sum += __shfl_xor(sum, off, 32);
    sq  += __shfl_xor(sq,  off, 32);
  }
  float mu  = sum * (1.0f / 128.0f);
  float var = sq * (1.0f / 128.0f) - mu * mu;
  float inv = rsqrtf(var + 1e-5f);
  const float4 sv = *(const float4*)(sc + lane * 4);
  const float4 bv = *(const float4*)(bi + lane * 4);
  bf16* yo = y + row * D_MODEL + lane * 4;
  yo[0] = f2bf((xv.x - mu) * inv * sv.x + bv.x);
  yo[1] = f2bf((xv.y - mu) * inv * sv.y + bv.y);
  yo[2] = f2bf((xv.z - mu) * inv * sv.z + bv.z);
  yo[3] = f2bf((xv.w - mu) * inv * sv.w + bv.w);
}

// ---------------------------------------------------------------------------
// WMMA GEMM:  out[M,OUT] = A[M,K](bf16) @ W[OUT,K]^T (bf16) + bias (+epilogue)
// One wave per 16x64 output tile: a single A fragment (16 rows x 32 K) is
// reused across 4 column sub-tiles per K-step -> 4x arithmetic intensity vs
// a 16x16 tile.  B-layout lanes read 16 contiguous K elements of a weight
// row (coalesced, since we need W^T).
// MODE: 0 = +bias -> bf16 out          (QKV projection)
//       1 = +bias +residual -> f32 out (Wo / FFN2)
//       2 = +bias, exact GELU -> bf16  (FFN1)
// ---------------------------------------------------------------------------
template<int K, int OUT, int MODE>
__global__ __launch_bounds__(256) void gemm_bf16_kernel(
    const bf16* __restrict__ A, const bf16* __restrict__ W,
    const float* __restrict__ bias, const float* __restrict__ res,
    void* __restrict__ outv, int M)
{
  const int wave = threadIdx.x >> 5;
  const int lane = threadIdx.x & 31;
  const int half = lane >> 4;
  const int lrow = lane & 15;
  const long tile = (long)blockIdx.x * 8 + wave;
  const int colGroups = OUT / 64;
  const long trow = tile / colGroups;
  const long tcol = tile % colGroups;
  if (trow * 16 >= M) return;

  const bf16* Ap = A + (trow * 16 + lrow) * (long)K + half * 16;
  const bf16* Wp0 = W + (tcol * 64 + lrow) * (long)K + half * 16;

  v8f acc[4] = {};
  #pragma unroll
  for (int ks = 0; ks < K; ks += 32) {
    union { v16bf v; uint4 u[2]; } a;
    a.u[0] = *(const uint4*)(Ap + ks);
    a.u[1] = *(const uint4*)(Ap + ks + 8);
    #pragma unroll
    for (int sub = 0; sub < 4; sub++) {
      union { v16bf v; uint4 u[2]; } b;
      const bf16* Wp = Wp0 + (long)sub * 16 * K;
      b.u[0] = *(const uint4*)(Wp + ks);
      b.u[1] = *(const uint4*)(Wp + ks + 8);
      acc[sub] = __builtin_amdgcn_wmma_f32_16x16x32_bf16(
          false, a.v, false, b.v, (short)0, acc[sub], false, false);
    }
  }

  #pragma unroll
  for (int sub = 0; sub < 4; sub++) {
    const long col = tcol * 64 + sub * 16 + lrow;
    const float bv = bias[col];
    #pragma unroll
    for (int i = 0; i < 8; i++) {
      const long row = trow * 16 + half * 8 + i;
      float vo = acc[sub][i] + bv;
      if constexpr (MODE == 1) {
        vo += res[row * OUT + col];
        ((float*)outv)[row * OUT + col] = vo;
      } else if constexpr (MODE == 2) {
        vo = 0.5f * vo * (1.0f + erff(vo * 0.70710678118654752f));
        ((bf16*)outv)[row * OUT + col] = f2bf(vo);
      } else {
        ((bf16*)outv)[row * OUT + col] = f2bf(vo);
      }
    }
  }
}

// ---------------------------------------------------------------------------
// Attention: one block (256 thr = 8 waves) per (segment, head).
// qkv bf16 layout [N, 384]: q at h*32, k at 128+h*32, v at 256+h*32.
// Each wave processes two 16-query tiles. Full 256-key scores held in
// 16 v8f accumulators; softmax via half-wave shuffles (1/sqrt(32) folded
// into exp); P converted C-layout -> A-layout through per-wave LDS scratch
// (DS ops are in-order within a wave, so only a compiler barrier is needed).
// ---------------------------------------------------------------------------
__global__ __launch_bounds__(256) void attn_kernel(
    const bf16* __restrict__ qkv, bf16* __restrict__ attn)
{
  __shared__ bf16 sQ[SEG * HEAD_D];      // [key_row][d]
  __shared__ bf16 sK[SEG * HEAD_D];      // [key_row][d]
  __shared__ bf16 sVt[HEAD_D * SEG];     // [d][key_row]  (transposed)
  __shared__ bf16 sP[8][16 * 32];        // per-wave prob scratch

  const int seg = blockIdx.x >> 2;
  const int h   = blockIdx.x & 3;
  const int tid = threadIdx.x;

  { // cooperative load: thread tid owns key row `tid`
    const long base = ((long)seg * SEG + tid) * (3 * D_MODEL) + h * HEAD_D;
    const bf16* q = qkv + base;
    const bf16* k = qkv + base + D_MODEL;
    const bf16* v = qkv + base + 2 * D_MODEL;
    #pragma unroll
    for (int c = 0; c < HEAD_D; c += 8) {
      *(uint4*)&sQ[tid * HEAD_D + c] = *(const uint4*)(q + c);
      *(uint4*)&sK[tid * HEAD_D + c] = *(const uint4*)(k + c);
      union { uint4 u; unsigned short s[8]; } vv;
      vv.u = *(const uint4*)(v + c);
      #pragma unroll
      for (int e = 0; e < 8; e++)
        sVt[(c + e) * SEG + tid] = __builtin_bit_cast(bf16, vv.s[e]);
    }
  }
  __syncthreads();

  const int wave = tid >> 5, lane = tid & 31;
  const int half = lane >> 4, lrow = lane & 15;
  const float scale = 0.17677669529663687f;  // 1/sqrt(32)

  for (int j = 0; j < 2; j++) {
    const int t = wave + j * 8;             // query tile: rows t*16 .. t*16+15
    union { v16bf v; uint4 u[2]; } qa;
    const int qrow = t * 16 + lrow;
    qa.u[0] = *(const uint4*)&sQ[qrow * HEAD_D + half * 16];
    qa.u[1] = *(const uint4*)&sQ[qrow * HEAD_D + half * 16 + 8];

    // ---- scores: 16 x (16x16) tiles over 256 keys --------------------
    v8f s[16];
    #pragma unroll
    for (int n = 0; n < 16; n++) {
      union { v16bf v; uint4 u[2]; } kb;
      const int key = n * 16 + lrow;
      kb.u[0] = *(const uint4*)&sK[key * HEAD_D + half * 16];
      kb.u[1] = *(const uint4*)&sK[key * HEAD_D + half * 16 + 8];
      v8f z = {};
      s[n] = __builtin_amdgcn_wmma_f32_16x16x32_bf16(
          false, qa.v, false, kb.v, (short)0, z, false, false);
    }

    // ---- softmax (row = half*8 + i, lives in 16-lane half) -----------
    float lsum[8];
    #pragma unroll
    for (int i = 0; i < 8; i++) {
      float mv = s[0][i];
      #pragma unroll
      for (int n = 1; n < 16; n++) mv = fmaxf(mv, s[n][i]);
      #pragma unroll
      for (int off = 8; off > 0; off >>= 1) mv = fmaxf(mv, __shfl_xor(mv, off, 32));
      float sum = 0.0f;
      #pragma unroll
      for (int n = 0; n < 16; n++) {
        float e = __expf((s[n][i] - mv) * scale);
        s[n][i] = e;
        sum += e;
      }
      #pragma unroll
      for (int off = 8; off > 0; off >>= 1) sum += __shfl_xor(sum, off, 32);
      lsum[i] = sum;
    }

    // ---- O = P @ V : 8 chunks of 32 keys -----------------------------
    v8f o0 = {}, o1 = {};
    #pragma unroll
    for (int cp = 0; cp < 8; cp++) {
      // C-layout -> row-major in LDS scratch
      #pragma unroll
      for (int cc = 0; cc < 2; cc++) {
        const int n = cp * 2 + cc;
        #pragma unroll
        for (int i = 0; i < 8; i++)
          sP[wave][(half * 8 + i) * 32 + cc * 16 + lrow] = f2bf(s[n][i]);
      }
      // DS ops are in-order within a wave; forbid compiler reordering only.
      asm volatile("" ::: "memory");
      union { v16bf v; uint4 u[2]; } pa, vb0, vb1;
      pa.u[0] = *(const uint4*)&sP[wave][lrow * 32 + half * 16];
      pa.u[1] = *(const uint4*)&sP[wave][lrow * 32 + half * 16 + 8];
      const int kbase = cp * 32 + half * 16;
      vb0.u[0] = *(const uint4*)&sVt[lrow * SEG + kbase];
      vb0.u[1] = *(const uint4*)&sVt[lrow * SEG + kbase + 8];
      vb1.u[0] = *(const uint4*)&sVt[(16 + lrow) * SEG + kbase];
      vb1.u[1] = *(const uint4*)&sVt[(16 + lrow) * SEG + kbase + 8];
      o0 = __builtin_amdgcn_wmma_f32_16x16x32_bf16(
          false, pa.v, false, vb0.v, (short)0, o0, false, false);
      o1 = __builtin_amdgcn_wmma_f32_16x16x32_bf16(
          false, pa.v, false, vb1.v, (short)0, o1, false, false);
    }

    // ---- normalize + store bf16 to attn[N,128] at head offset --------
    #pragma unroll
    for (int i = 0; i < 8; i++) {
      const float inv = 1.0f / lsum[i];
      const long row = (long)seg * SEG + t * 16 + half * 8 + i;
      attn[row * D_MODEL + h * HEAD_D + lrow]      = f2bf(o0[i] * inv);
      attn[row * D_MODEL + h * HEAD_D + 16 + lrow] = f2bf(o1[i] * inv);
    }
  }
}

// ---------------------------------------------------------------------------
// Host-side orchestration
// ---------------------------------------------------------------------------
static inline size_t align256(size_t v) { return (v + 255) & ~(size_t)255; }

extern "C" void kernel_launch(void* const* d_in, const int* in_sizes, int n_in,
                              void* d_out, int out_size, void* d_ws, size_t ws_size,
                              hipStream_t stream) {
  const int   N        = in_sizes[0];
  const int*  toks     = (const int*)d_in[0];
  const int*  mask_id  = (const int*)d_in[2];
  const float* tok_emb = (const float*)d_in[4];
  const float* pos_emb = (const float*)d_in[5];
  const float* mask_emb= (const float*)d_in[6];
  const float* Wqkv    = (const float*)d_in[7];
  const float* bqkv    = (const float*)d_in[8];
  const float* Wo      = (const float*)d_in[9];
  const float* bo      = (const float*)d_in[10];
  const float* W1      = (const float*)d_in[11];
  const float* b1      = (const float*)d_in[12];
  const float* W2      = (const float*)d_in[13];
  const float* b2      = (const float*)d_in[14];
  const float* ln1_s   = (const float*)d_in[15];
  const float* ln1_b   = (const float*)d_in[16];
  const float* ln2_s   = (const float*)d_in[17];
  const float* ln2_b   = (const float*)d_in[18];
  float* out = (float*)d_out;

  // workspace carve-up
  char* ws = (char*)d_ws;
  float* x    = (float*)ws;  ws += align256((size_t)N * D_MODEL * 4);
  bf16*  y    = (bf16*)ws;   ws += align256((size_t)N * D_MODEL * 2);
  bf16*  qkvb = (bf16*)ws;   ws += align256((size_t)N * 3 * D_MODEL * 2);
  bf16*  attnb= (bf16*)ws;   ws += align256((size_t)N * D_MODEL * 2);
  bf16*  hb   = (bf16*)ws;   ws += align256((size_t)N * FF_DIM * 2);
  bf16*  wqkvb= (bf16*)ws;   ws += align256((size_t)N_LAYERS * 3 * D_MODEL * D_MODEL * 2);
  bf16*  wob  = (bf16*)ws;   ws += align256((size_t)N_LAYERS * D_MODEL * D_MODEL * 2);
  bf16*  w1b  = (bf16*)ws;   ws += align256((size_t)N_LAYERS * FF_DIM * D_MODEL * 2);
  bf16*  w2b  = (bf16*)ws;   ws += align256((size_t)N_LAYERS * D_MODEL * FF_DIM * 2);

  // weights -> bf16
  {
    const int nqkv = N_LAYERS * 3 * D_MODEL * D_MODEL;
    const int nwo  = N_LAYERS * D_MODEL * D_MODEL;
    const int nw1  = N_LAYERS * FF_DIM * D_MODEL;
    const int nw2  = N_LAYERS * D_MODEL * FF_DIM;
    cvt_bf16_kernel<<<(nqkv + 255) / 256, 256, 0, stream>>>(Wqkv, wqkvb, nqkv);
    cvt_bf16_kernel<<<(nwo  + 255) / 256, 256, 0, stream>>>(Wo,   wob,  nwo);
    cvt_bf16_kernel<<<(nw1  + 255) / 256, 256, 0, stream>>>(W1,   w1b,  nw1);
    cvt_bf16_kernel<<<(nw2  + 255) / 256, 256, 0, stream>>>(W2,   w2b,  nw2);
  }

  // embeddings
  embed_kernel<<<(N * 32 + 255) / 256, 256, 0, stream>>>(
      toks, mask_id, tok_emb, pos_emb, mask_emb, x, N);

  const int rowTiles = N / 16;
  const int lnBlocks = (N + 7) / 8;

  for (int l = 0; l < N_LAYERS; l++) {
    // y = LN1(x)
    ln_kernel<<<lnBlocks, 256, 0, stream>>>(
        x, ln1_s + l * D_MODEL, ln1_b + l * D_MODEL, y, N);
    // qkv = y @ Wqkv^T + bqkv  -> bf16
    gemm_bf16_kernel<D_MODEL, 3 * D_MODEL, 0>
        <<<rowTiles * (3 * D_MODEL / 64) / 8, 256, 0, stream>>>(
        y, wqkvb + (size_t)l * 3 * D_MODEL * D_MODEL, bqkv + l * 3 * D_MODEL,
        nullptr, qkvb, N);
    // attention -> bf16 [N,128]
    attn_kernel<<<(N / SEG) * N_HEADS, 256, 0, stream>>>(qkvb, attnb);
    // x = x + attn @ Wo^T + bo
    gemm_bf16_kernel<D_MODEL, D_MODEL, 1>
        <<<rowTiles * (D_MODEL / 64) / 8, 256, 0, stream>>>(
        attnb, wob + (size_t)l * D_MODEL * D_MODEL, bo + l * D_MODEL, x, x, N);
    // y = LN2(x)
    ln_kernel<<<lnBlocks, 256, 0, stream>>>(
        x, ln2_s + l * D_MODEL, ln2_b + l * D_MODEL, y, N);
    // h = gelu(y @ W1^T + b1) -> bf16
    gemm_bf16_kernel<D_MODEL, FF_DIM, 2>
        <<<rowTiles * (FF_DIM / 64) / 8, 256, 0, stream>>>(
        y, w1b + (size_t)l * FF_DIM * D_MODEL, b1 + l * FF_DIM, nullptr, hb, N);
    // x' = x + h @ W2^T + b2   (last layer writes d_out)
    float* dst = (l == N_LAYERS - 1) ? out : x;
    gemm_bf16_kernel<FF_DIM, D_MODEL, 1>
        <<<rowTiles * (D_MODEL / 64) / 8, 256, 0, stream>>>(
        hb, w2b + (size_t)l * D_MODEL * FF_DIM, b2 + l * D_MODEL, x, dst, N);
  }
}
